// Net_Actor_44890998178491
// MI455X (gfx1250) — matrix operand order repaired and checked by reading, not compiled, not run on hardware
//
#include <hip/hip_runtime.h>
#include <hip/hip_bf16.h>
#include <math.h>

typedef __bf16 bf16_t;
typedef __attribute__((ext_vector_type(16))) __bf16 v16bf;
typedef __attribute__((ext_vector_type(8)))  __bf16 v8bf;
typedef __attribute__((ext_vector_type(8)))  float  v8f;
typedef __attribute__((ext_vector_type(4)))  unsigned int uint32x4;
typedef __attribute__((ext_vector_type(8)))  int int32x8;
typedef __attribute__((ext_vector_type(4)))  int int32x4;

#define TPB 256

// ---------------------------------------------------------------- utilities

__global__ void cvt_bf16_kernel(const float* __restrict__ src, bf16_t* __restrict__ dst, int n) {
    int i = blockIdx.x * blockDim.x + threadIdx.x;
    if (i < n) dst[i] = (bf16_t)src[i];
}

__device__ __forceinline__ float lrelu(float x) { return x > 0.f ? x : 0.2f * x; }

__device__ __forceinline__ void atomicMaxF(float* addr, float v) {
    if (v >= 0.f) atomicMax((int*)addr, __float_as_int(v));
    else          atomicMin((unsigned int*)addr, __float_as_uint(v));
}

__device__ __forceinline__ v16bf pack16(v8bf lo, v8bf hi) {
    v16bf r;
#pragma unroll
    for (int i = 0; i < 8; ++i) { r[i] = lo[i]; r[i + 8] = hi[i]; }
    return r;
}

// ---------------------------------------------------------------- WMMA compute core
// Block = 256 threads = 8 waves covers 64 rows x 128 cols.
// Wave w -> cols [16w,16w+16), 4 independent row-tile accumulators sharing one B frag.
__device__ __forceinline__ void gemm_core(const bf16_t* As, int LSTR,
                                          const bf16_t* __restrict__ W, int ldw,
                                          const float* __restrict__ bias,
                                          float* __restrict__ C, int ldc,
                                          int K, int act, int row0) {
    const int tid  = threadIdx.x;
    const int wave = tid >> 5;
    const int lane = tid & 31;
    const int half = lane >> 4;
    const int l16  = lane & 15;
    const int col0 = wave * 16;

    v8f acc[4] = {};
    for (int kt = 0; kt < K; kt += 32) {
        // B fragment: B(k,n)=W[col0+n][k]; lane n=l16: elems 0-7 k=kt+half*8+e, 8-15 k=kt+16+half*8+e
        const bf16_t* bp = &W[(size_t)(col0 + l16) * ldw + kt + half * 8];
        v16bf bfr = pack16(*(const v8bf*)bp, *(const v8bf*)(bp + 16));
#pragma unroll
        for (int r = 0; r < 4; ++r) {
            // A fragment (ISA 16-bit A 16x32): elems 0-7 k=kt+half*8+e, 8-15 k=kt+16+half*8+e
            const bf16_t* ap = &As[(r * 16 + l16) * LSTR + kt + half * 8];
            v16bf af = pack16(*(const v8bf*)ap, *(const v8bf*)(ap + 16));
            acc[r] = __builtin_amdgcn_wmma_f32_16x16x32_bf16(false, af, false, bfr,
                                                             (short)0, acc[r], false, false);
        }
    }
    // C/D layout: lane -> n=l16 ; VGPR j -> m = j + 8*half
#pragma unroll
    for (int r = 0; r < 4; ++r) {
#pragma unroll
        for (int j = 0; j < 8; ++j) {
            int m = r * 16 + j + 8 * half;
            int n = col0 + l16;
            float v = acc[r][j] + (bias ? bias[n] : 0.f);
            if (act == 1)      v = 1.f / (1.f + __expf(-v));   // sigmoid
            else if (act == 2) v = tanhf(v);                   // tanh
            C[(size_t)(row0 + m) * ldc + n] = v;
        }
    }
}

// fp32-A variant (VALU convert staging) — used for the small GRU GEMMs (K up to 256).
__global__ void gemm_wmma_f32a(const float* __restrict__ A, int lda,
                               const bf16_t* __restrict__ W, int ldw,
                               const float* __restrict__ bias,
                               float* __restrict__ C, int ldc,
                               int K, int act) {
    __shared__ __align__(16) bf16_t As[64 * (256 + 8)];
    const int LSTR = K + 8;
    const int row0 = blockIdx.x * 64;
    const int tot  = 64 * K;
    for (int idx = threadIdx.x; idx < tot; idx += TPB) {
        int r = idx / K, c = idx - r * K;
        As[r * LSTR + c] = (bf16_t)A[(size_t)(row0 + r) * lda + c];
    }
    __syncthreads();
    gemm_core(As, LSTR, W, ldw, bias, C, ldc, K, act, row0);
}

// bf16-A variant staged by the Tensor Data Mover (K = 128).
// D# (ISA ch.8): count=1, type=2, data_size=2B, 2D tile K x 64 rows, stride=lda elems,
// LDS pad: 4 DWORDs every K/2 DWORDs (reproduces the K+8 element row stride).
__global__ void gemm_wmma_tdm(const bf16_t* __restrict__ A, int lda,
                              const bf16_t* __restrict__ W, int ldw,
                              const float* __restrict__ bias,
                              float* __restrict__ C, int ldc,
                              int K, int act) {
    __shared__ __align__(16) bf16_t As[64 * (128 + 8)];
    const int LSTR = K + 8;
    const int row0 = blockIdx.x * 64;
    if (threadIdx.x < 32) {   // wave 0 issues the DMA (TDM ignores EXEC; one op per wave)
        unsigned long long gaddr = (unsigned long long)(uintptr_t)(A + (size_t)row0 * lda);
        unsigned int laddr = (unsigned int)(uintptr_t)&As[0];   // LDS aperture: low 32 bits
        int icode = 0; { int d = K / 2; while ((2 << icode) < d) ++icode; } // (2<<icode)==K/2 DW
        uint32x4 g0;
        g0[0] = 1u;                                            // count=1, user mode
        g0[1] = laddr;                                         // lds_addr (bytes)
        g0[2] = (unsigned int)gaddr;                           // global_addr[31:0]
        g0[3] = (unsigned int)((gaddr >> 32) & 0x1FFFFFFull) | (2u << 30); // [56:32] | type=2
        int32x8 g1;
        g1[0] = (1 << 16)        // data_size = 1 -> 2 bytes
              | (1 << 20)        // pad_enable
              | (icode << 22)    // pad_interval: K/2 DWORDs
              | (3 << 25);       // pad_amount: 4 DWORDs (16B)
        g1[1] = (K & 0xFFFF) << 16;                            // tensor_dim0[15:0]
        g1[2] = ((K >> 16) & 0xFFFF) | (64 << 16);             // tensor_dim0 hi | tensor_dim1 lo
        g1[3] = (K & 0xFFFF) << 16;                            // tensor_dim1 hi=0 | tile_dim0=K
        g1[4] = 64;                                            // tile_dim1=64 rows, tile_dim2=0
        g1[5] = lda;                                           // tensor_dim0_stride[31:0] (elems)
        g1[6] = 0;                                             // stride hi | dim1_stride lo
        g1[7] = 0;
        int32x4 z4 = {0, 0, 0, 0};
        int32x8 z8 = {0, 0, 0, 0, 0, 0, 0, 0};
        __builtin_amdgcn_tensor_load_to_lds(g0, g1, z4, z4, z8, 0);
        __builtin_amdgcn_s_wait_tensorcnt(0);
    }
    __syncthreads();
    gemm_core(As, LSTR, W, ldw, bias, C, ldc, K, act, row0);
}

// ---------------------------------------------------------------- GRU glue

__global__ void mean_m_kernel(const float* __restrict__ inp, float* __restrict__ im,
                              int Bn, int Mn) {
    int i = blockIdx.x * blockDim.x + threadIdx.x;
    if (i >= Bn * 128) return;
    int b = i >> 7, c = i & 127;
    float s = 0.f;
    for (int m = 0; m < Mn; ++m) s += inp[((size_t)b * Mn + m) * 128 + c];
    im[i] = s / (float)Mn;
}

__global__ void copy_state_si(const float* __restrict__ st, float* __restrict__ si, int Bn) {
    int i = blockIdx.x * blockDim.x + threadIdx.x;
    if (i >= Bn * 128) return;
    int b = i >> 7, c = i & 127;
    si[(size_t)b * 256 + c] = st[i];
}

__global__ void build_rsi(const float* __restrict__ r, const float* __restrict__ st,
                          const float* __restrict__ si, float* __restrict__ rsi, int Bn) {
    int i = blockIdx.x * blockDim.x + threadIdx.x;
    if (i >= Bn * 128) return;
    int b = i >> 7, c = i & 127;
    rsi[(size_t)b * 256 + c]       = r[i] * st[i];
    rsi[(size_t)b * 256 + 128 + c] = si[(size_t)b * 256 + 128 + c];
}

__global__ void gru_out_kernel(const float* __restrict__ z, const float* __restrict__ st,
                               const float* __restrict__ hc, float* __restrict__ outp, int n) {
    int i = blockIdx.x * blockDim.x + threadIdx.x;
    if (i < n) outp[i] = (1.f - z[i]) * st[i] + z[i] * hc[i];
}

// ---------------------------------------------------------------- GAT edge phase

__global__ void gat_init(float* __restrict__ amax, float* __restrict__ den,
                         float* __restrict__ outp, const float* __restrict__ res, int n) {
    int i = blockIdx.x * blockDim.x + threadIdx.x;
    if (i >= n) return;
    amax[i] = -INFINITY;
    den[i]  = 0.f;
    outp[i] = res ? res[i] : 0.f;
}

__global__ void gat_edge_max(const int* __restrict__ src, const int* __restrict__ dst,
                             int E, int Nn, const float* __restrict__ u,
                             const float* __restrict__ v, float* __restrict__ amax) {
    int idx = blockIdx.x * blockDim.x + threadIdx.x;
    int tot = (E + Nn) * 32;
    if (idx >= tot) return;
    int e = idx >> 5, g = (idx & 31) * 4;
    int s, d;
    if (e < E) { s = src[e]; d = dst[e]; } else { s = d = e - E; }
    const float4 ud = *(const float4*)(u + (size_t)d * 128 + g);
    const float4 vs = *(const float4*)(v + (size_t)s * 128 + g);
    float av[4] = { ud.x + vs.x, ud.y + vs.y, ud.z + vs.z, ud.w + vs.w };
#pragma unroll
    for (int i = 0; i < 4; ++i) atomicMaxF(&amax[(size_t)d * 128 + g + i], lrelu(av[i]));
}

__global__ void gat_edge_sum(const int* __restrict__ src, const int* __restrict__ dst,
                             int E, int Nn, const float* __restrict__ u,
                             const float* __restrict__ v, const float* __restrict__ amax,
                             float* __restrict__ den) {
    int idx = blockIdx.x * blockDim.x + threadIdx.x;
    int tot = (E + Nn) * 32;
    if (idx >= tot) return;
    int e = idx >> 5, g = (idx & 31) * 4;
    int s, d;
    if (e < E) { s = src[e]; d = dst[e]; } else { s = d = e - E; }
    const float4 ud = *(const float4*)(u + (size_t)d * 128 + g);
    const float4 vs = *(const float4*)(v + (size_t)s * 128 + g);
    const float4 am = *(const float4*)(amax + (size_t)d * 128 + g);
    float av[4] = { ud.x + vs.x, ud.y + vs.y, ud.z + vs.z, ud.w + vs.w };
    float mm[4] = { am.x, am.y, am.z, am.w };
#pragma unroll
    for (int i = 0; i < 4; ++i)
        atomicAdd(&den[(size_t)d * 128 + g + i], __expf(lrelu(av[i]) - mm[i]));
}

__global__ void gat_edge_out(const int* __restrict__ src, const int* __restrict__ dst,
                             int E, int Nn, const float* __restrict__ u,
                             const float* __restrict__ v, const float* __restrict__ h,
                             const float* __restrict__ amax, const float* __restrict__ den,
                             float* __restrict__ outp) {
    int idx = blockIdx.x * blockDim.x + threadIdx.x;
    int tot = (E + Nn) * 32;
    if (idx >= tot) return;
    int e = idx >> 5, g = (idx & 31) * 4;
    int s, d;
    if (e < E) { s = src[e]; d = dst[e]; } else { s = d = e - E; }
    const float4 ud = *(const float4*)(u + (size_t)d * 128 + g);
    const float4 vs = *(const float4*)(v + (size_t)s * 128 + g);
    const float4 am = *(const float4*)(amax + (size_t)d * 128 + g);
    const float4 dn = *(const float4*)(den + (size_t)d * 128 + g);
    const float4 hs = *(const float4*)(h + (size_t)s * 128 + g);
    float av[4] = { ud.x + vs.x, ud.y + vs.y, ud.z + vs.z, ud.w + vs.w };
    float mm[4] = { am.x, am.y, am.z, am.w };
    float dd[4] = { dn.x, dn.y, dn.z, dn.w };
    float hh[4] = { hs.x, hs.y, hs.z, hs.w };
#pragma unroll
    for (int i = 0; i < 4; ++i) {
        float alpha = __expf(lrelu(av[i]) - mm[i]) / (dd[i] + 1e-16f);
        atomicAdd(&outp[(size_t)d * 128 + g + i], hh[i] * alpha);
    }
}

// ---------------------------------------------------------------- heads

__global__ void heads_kernel(const float* __restrict__ gx, int NNm,
                             const float* __restrict__ pw, const float* __restrict__ pb,
                             const float* __restrict__ mw, const float* __restrict__ mb,
                             const float* __restrict__ sw, const float* __restrict__ sb,
                             float* __restrict__ scores, float* __restrict__ a_out,
                             float* __restrict__ b_out, int Bn) {
    int gw   = (blockIdx.x * blockDim.x + threadIdx.x) >> 5;
    int lane = threadIdx.x & 31;
    int rows = NNm - 1;
    if (gw >= Bn * rows) return;
    int b = gw / rows, j = gw - b * rows + 1;
    const float* xp = gx + ((size_t)b * NNm + j) * 128;
    float ps = 0.f, ms = 0.f, ss = 0.f;
    for (int c = lane; c < 128; c += 32) {
        float xv = xp[c];
        ps += xv * pw[c]; ms += xv * mw[c]; ss += xv * sw[c];
    }
#pragma unroll
    for (int off = 16; off > 0; off >>= 1) {
        ps += __shfl_xor(ps, off, 32);
        ms += __shfl_xor(ms, off, 32);
        ss += __shfl_xor(ss, off, 32);
    }
    if (lane == 0) {
        int oi = b * rows + (j - 1);
        scores[oi] = ps + pb[0];
        float mv = ms + mb[0];
        a_out[oi] = (mv > 0.f ? mv : __expf(mv) - 1.f) + 2.f;
        float sv = fabsf(ss + sb[0]);
        b_out[oi] = (sv > 0.f ? sv : __expf(sv) - 1.f) + 2.f;
    }
}

__global__ void softmax_rows(const float* __restrict__ scores, float* __restrict__ probs, int L) {
    __shared__ float red[TPB];
    int b = blockIdx.x, t = threadIdx.x;
    const float* s = scores + (size_t)b * L;
    float mx = -INFINITY;
    for (int i = t; i < L; i += TPB) mx = fmaxf(mx, s[i]);
    red[t] = mx; __syncthreads();
    for (int o = TPB / 2; o > 0; o >>= 1) { if (t < o) red[t] = fmaxf(red[t], red[t + o]); __syncthreads(); }
    mx = red[0]; __syncthreads();
    float sm = 0.f;
    for (int i = t; i < L; i += TPB) sm += __expf(s[i] - mx);
    red[t] = sm; __syncthreads();
    for (int o = TPB / 2; o > 0; o >>= 1) { if (t < o) red[t] += red[t + o]; __syncthreads(); }
    sm = red[0];
    for (int i = t; i < L; i += TPB) probs[(size_t)b * L + i] = __expf(s[i] - mx) / sm;
}

// ---------------------------------------------------------------- host driver

static inline int cdiv(int a, int b) { return (a + b - 1) / b; }

extern "C" void kernel_launch(void* const* d_in, const int* in_sizes, int n_in,
                              void* d_out, int out_size, void* d_ws, size_t ws_size,
                              hipStream_t stream) {
    const int Bn = in_sizes[0] / 128;          // 64
    const int Nn = in_sizes[2] / 128;          // 32000
    const int E  = in_sizes[3] / 2;            // 256000
    const int Mm = in_sizes[1] / (Bn * 128);   // 20
    const int NNm = Nn / Bn;                   // 500

    const float* state_ = (const float*)d_in[0];
    const float* input_ = (const float*)d_in[1];
    const float* x      = (const float*)d_in[2];
    const int*   e_n    = (const int*)d_in[3];
    const int*   e_r0   = (const int*)d_in[4];
    const int*   e_r1   = (const int*)d_in[5];
    const float* win = (const float*)d_in[7];
    const float* wz = (const float*)d_in[8];   const float* bz = (const float*)d_in[9];
    const float* wr = (const float*)d_in[10];  const float* br = (const float*)d_in[11];
    const float* wh = (const float*)d_in[12];  const float* bh = (const float*)d_in[13];
    const float* g0_lw = (const float*)d_in[14]; const float* g0_lb = (const float*)d_in[15];
    const float* g0_aw = (const float*)d_in[16]; const float* g0_ab = (const float*)d_in[17];
    const float* g1_lw = (const float*)d_in[18]; const float* g1_lb = (const float*)d_in[19];
    const float* g1_aw = (const float*)d_in[20]; const float* g1_ab = (const float*)d_in[21];
    const float* g2_lw = (const float*)d_in[22]; const float* g2_lb = (const float*)d_in[23];
    const float* g2_aw = (const float*)d_in[24]; const float* g2_ab = (const float*)d_in[25];
    const float* gn_lw = (const float*)d_in[26]; const float* gn_lb = (const float*)d_in[27];
    const float* gn_aw = (const float*)d_in[28]; const float* gn_ab = (const float*)d_in[29];
    const float* pw = (const float*)d_in[30]; const float* pb = (const float*)d_in[31];
    const float* mw = (const float*)d_in[32]; const float* mb = (const float*)d_in[33];
    const float* sw = (const float*)d_in[34]; const float* sb = (const float*)d_in[35];

    // -------- workspace carve-out (256B aligned)
    char* ws = (char*)d_ws;
    size_t off = 0;
    auto alloc = [&](size_t bytes) -> char* {
        char* p = ws + off;
        off += (bytes + 255) & ~(size_t)255;
        return p;
    };
    bf16_t* winb  = (bf16_t*)alloc(128 * 128 * 2);
    bf16_t* wzb   = (bf16_t*)alloc(128 * 256 * 2);
    bf16_t* wrb   = (bf16_t*)alloc(128 * 256 * 2);
    bf16_t* whb   = (bf16_t*)alloc(128 * 256 * 2);
    bf16_t* g0lwb = (bf16_t*)alloc(128 * 128 * 2);
    bf16_t* g0awb = (bf16_t*)alloc(128 * 256 * 2);
    bf16_t* g1lwb = (bf16_t*)alloc(128 * 128 * 2);
    bf16_t* g1awb = (bf16_t*)alloc(128 * 256 * 2);
    bf16_t* g2lwb = (bf16_t*)alloc(128 * 128 * 2);
    bf16_t* g2awb = (bf16_t*)alloc(128 * 256 * 2);
    bf16_t* gnlwb = (bf16_t*)alloc(128 * 128 * 2);
    bf16_t* gnawb = (bf16_t*)alloc(128 * 256 * 2);

    const size_t NF = (size_t)Nn * 128;
    float* bufA = (float*)alloc(NF * 4);
    float* bufB = (float*)alloc(NF * 4);
    float* bufC = (float*)alloc(NF * 4);
    float* hbuf = (float*)alloc(NF * 4);
    float* ubuf = (float*)alloc(NF * 4);
    float* vbuf = (float*)alloc(NF * 4);
    float* amaxb = (float*)alloc(NF * 4);
    float* denb  = (float*)alloc(NF * 4);
    bf16_t* xbf  = (bf16_t*)alloc(NF * 2);   // bf16 activations for TDM-fed GEMMs
    bf16_t* hbf  = (bf16_t*)alloc(NF * 2);

    float* im  = (float*)alloc((size_t)Bn * 128 * 4);
    float* si  = (float*)alloc((size_t)Bn * 256 * 4);
    float* zb  = (float*)alloc((size_t)Bn * 128 * 4);
    float* rb  = (float*)alloc((size_t)Bn * 128 * 4);
    float* rsi = (float*)alloc((size_t)Bn * 256 * 4);
    float* hcb = (float*)alloc((size_t)Bn * 128 * 4);
    float* scores = (float*)alloc((size_t)Bn * (NNm - 1) * 4);
    (void)ws_size; (void)n_in; (void)out_size;

    // -------- convert GEMM weights to bf16
    auto cvt = [&](const float* s, bf16_t* d, int n) {
        cvt_bf16_kernel<<<cdiv(n, TPB), TPB, 0, stream>>>(s, d, n);
    };
    cvt(win, winb, 128 * 128);   cvt(wz, wzb, 128 * 256);
    cvt(wr, wrb, 128 * 256);     cvt(wh, whb, 128 * 256);
    cvt(g0_lw, g0lwb, 128 * 128); cvt(g0_aw, g0awb, 128 * 256);
    cvt(g1_lw, g1lwb, 128 * 128); cvt(g1_aw, g1awb, 128 * 256);
    cvt(g2_lw, g2lwb, 128 * 128); cvt(g2_aw, g2awb, 128 * 256);
    cvt(gn_lw, gnlwb, 128 * 128); cvt(gn_aw, gnawb, 128 * 256);

    auto gemm_f = [&](const float* A, int lda, const bf16_t* W, int ldw,
                      const float* bias, float* C, int ldc, int M, int K, int act) {
        gemm_wmma_f32a<<<M / 64, TPB, 0, stream>>>(A, lda, W, ldw, bias, C, ldc, K, act);
    };
    auto gemm_t = [&](const bf16_t* A, int lda, const bf16_t* W, int ldw,
                      const float* bias, float* C, int ldc, int M, int K, int act) {
        gemm_wmma_tdm<<<M / 64, TPB, 0, stream>>>(A, lda, W, ldw, bias, C, ldc, K, act);
    };

    // -------- GRU head (M = 64 -> one 64-row block per GEMM)
    mean_m_kernel<<<cdiv(Bn * 128, TPB), TPB, 0, stream>>>(input_, im, Bn, Mm);
    copy_state_si<<<cdiv(Bn * 128, TPB), TPB, 0, stream>>>(state_, si, Bn);
    gemm_f(im, 128, winb, 128, nullptr, si + 128, 256, Bn, 128, 0);   // i -> si[:,128:]
    gemm_f(si, 256, wzb, 256, bz, zb, 128, Bn, 256, 1);               // z = sigmoid
    gemm_f(si, 256, wrb, 256, br, rb, 128, Bn, 256, 1);               // r = sigmoid
    build_rsi<<<cdiv(Bn * 128, TPB), TPB, 0, stream>>>(rb, state_, si, rsi, Bn);
    gemm_f(rsi, 256, whb, 256, bh, hcb, 128, Bn, 256, 2);             // hc = tanh
    gru_out_kernel<<<cdiv(Bn * 128, TPB), TPB, 0, stream>>>(zb, state_, hcb,
                                                            (float*)d_out, Bn * 128);

    // -------- GAT conv (src/dst decomposition, TDM-fed WMMA GEMMs)
    const int edgeBlocks = cdiv((E + Nn) * 32, TPB);
    const int nodeBlocks = cdiv((int)NF, TPB);
    auto gat = [&](const float* xin, const int* edges, const bf16_t* lwb, const float* lb,
                   const bf16_t* awb, const float* ab, const float* res, float* outp) {
        const int* srcp = edges;
        const int* dstp = edges + E;
        cvt_bf16_kernel<<<nodeBlocks, TPB, 0, stream>>>(xin, xbf, (int)NF);
        gemm_t(xbf, 128, lwb, 128, lb, hbuf, 128, Nn, 128, 0);            // h
        cvt_bf16_kernel<<<nodeBlocks, TPB, 0, stream>>>(hbuf, hbf, (int)NF);
        gemm_t(hbf, 128, awb, 256, ab, ubuf, 128, Nn, 128, 0);            // u = h@aw_i^T + ab
        gemm_t(hbf, 128, awb + 128, 256, nullptr, vbuf, 128, Nn, 128, 0); // v = h@aw_j^T
        gat_init<<<nodeBlocks, TPB, 0, stream>>>(amaxb, denb, outp, res, (int)NF);
        gat_edge_max<<<edgeBlocks, TPB, 0, stream>>>(srcp, dstp, E, Nn, ubuf, vbuf, amaxb);
        gat_edge_sum<<<edgeBlocks, TPB, 0, stream>>>(srcp, dstp, E, Nn, ubuf, vbuf, amaxb, denb);
        gat_edge_out<<<edgeBlocks, TPB, 0, stream>>>(srcp, dstp, E, Nn, ubuf, vbuf, hbuf,
                                                     amaxb, denb, outp);
    };

    gat(x,    e_n,  g0lwb, g0_lb, g0awb, g0_ab, nullptr, bufA);  // g0
    gat(bufA, e_r1, g1lwb, g1_lb, g1awb, g1_ab, nullptr, bufB);  // g1
    gat(bufB, e_r1, g1lwb, g1_lb, g1awb, g1_ab, nullptr, bufB);  // g2 (in-place ok)
    gat(bufB, e_n,  gnlwb, gn_lb, gnawb, gn_ab, bufA,    bufC);  // g3 = conv + g0
    gat(bufC, e_r0, g2lwb, g2_lb, g2awb, g2_ab, nullptr, bufB);  // g4
    gat(bufB, e_r0, g2lwb, g2_lb, g2awb, g2_ab, nullptr, bufB);  // g5
    gat(bufB, e_n,  gnlwb, gn_lb, gnawb, gn_ab, bufC,    bufA);  // gx = conv + g3

    // -------- heads: scores/a/b then softmax
    const int rows = NNm - 1;
    float* out_probs = (float*)d_out + (size_t)Bn * 128;
    float* out_a     = out_probs + (size_t)Bn * rows;
    float* out_b     = out_a + (size_t)Bn * rows;
    heads_kernel<<<cdiv(Bn * rows * 32, TPB), TPB, 0, stream>>>(bufA, NNm, pw, pb, mw, mb,
                                                                sw, sb, scores, out_a, out_b, Bn);
    softmax_rows<<<Bn, TPB, 0, stream>>>(scores, out_probs, rows);
}
